// SPCNNClassifier_1314259992790
// MI455X (gfx1250) — matrix operand order repaired and checked by reading, not compiled, not run on hardware
//
#include <hip/hip_runtime.h>
#include <hip/hip_bf16.h>

typedef __attribute__((ext_vector_type(16))) _Float16 v16h;
typedef __attribute__((ext_vector_type(8)))  _Float16 v8h;
typedef __attribute__((ext_vector_type(8)))  float    v8f;
typedef __attribute__((ext_vector_type(4)))  unsigned int v4u;
typedef __attribute__((ext_vector_type(8)))  int      v8i_;
typedef __attribute__((ext_vector_type(4)))  int      v4i_;

#define BB  256
#define TT  128
#define DD  1024
#define CFF 128
#define HH  128
#define KK  9

#if defined(__has_builtin)
#  if __has_builtin(__builtin_amdgcn_tensor_load_to_lds)
#    define HAS_TDM 1
#  endif
#endif
#ifndef HAS_TDM
#  define HAS_TDM 0
#endif

__device__ __forceinline__ float sigmoidf_(float x) { return 1.0f / (1.0f + expf(-x)); }
__device__ __forceinline__ float gelu_(float x) { return 0.5f * x * (1.0f + erff(x * 0.70710678118654752f)); }

#if HAS_TDM
// Issue one TDM 2D tile load: 64 rows (n) x 32 cols (k) of f16 from a [Npad][K]
// tensor into LDS (row-major [64][32]). Caller must s_wait_tensorcnt + barrier.
__device__ __forceinline__ void tdm_load_b_tile(const _Float16* gsrc, unsigned ldsDst,
                                                int K, int Npad)
{
  unsigned long long ga = (unsigned long long)(size_t)gsrc;
  v4u g0;
  g0.x = 1u;                                          // count=1, user mode
  g0.y = ldsDst;                                      // lds_addr
  g0.z = (unsigned)(ga & 0xffffffffull);              // global_addr[31:0]
  g0.w = (unsigned)((ga >> 32) & 0x1ffffffull) | (2u << 30);   // addr[56:32] | type=2
  v8i_ g1;
  g1[0] = 0x00010000;                                 // data_size=1 (2 bytes)
  g1[1] = (int)(((unsigned)K & 0xffffu) << 16);       // tensor_dim0 lo16 -> [31:16]
  g1[2] = (int)((((unsigned)K >> 16) & 0xffffu) | (((unsigned)Npad & 0xffffu) << 16));
  g1[3] = (int)((((unsigned)Npad >> 16) & 0xffffu) | (32u << 16));  // tile_dim0=32
  g1[4] = 64;                                         // tile_dim1=64, tile_dim2=0
  g1[5] = (int)K;                                     // tensor_dim0_stride lo32
  g1[6] = 0;
  g1[7] = 0;
  v4i_ gz = {};
#if defined(__clang_major__) && __clang_major__ >= 23
  v8i_ gz8 = {};
  __builtin_amdgcn_tensor_load_to_lds(g0, g1, gz, gz, gz8, 0);
#else
  __builtin_amdgcn_tensor_load_to_lds(g0, g1, gz, gz, 0);
#endif
}
#endif

// ---------------------------------------------------------------------------
// Block-tiled WMMA conv/GEMM, f16 operands, f32 accumulate.
//   Y[row, colOff+n] = bias(+bias2) + sum_tap sum_k A[(b*T + t+tap-pad), k] * Bw[tap][n][k]
// Block = 8 waves -> 128(M) x 64(N); wave = 16(M) x 64(N), 4 v8f accumulators.
// B tile (32k x 64n, f16) staged into LDS via TDM, double-buffered.
// ---------------------------------------------------------------------------
__global__ void __launch_bounds__(256)
wmma_gemm_tdm(const _Float16* __restrict__ A,
              const _Float16* __restrict__ Bw,    // [taps][Npad][K] f16 (zero padded)
              const float* __restrict__ bias,
              const float* __restrict__ bias2,
              float* __restrict__ Y,
              int T, int K, int taps, int pad,
              int Nvalid, int nBlocks, int ldy, int colOff, int Npad)
{
  __shared__ __align__(16) _Float16 bt[2][64 * 32];
  int wv   = threadIdx.x >> 5;
  int lane = threadIdx.x & 31;
  int mB = blockIdx.x / nBlocks;
  int nB = blockIdx.x - mB * nBlocks;
  int m0 = mB * 128 + wv * 16;
  int n0 = nB * 64;

  int rowm = m0 + (lane & 15);
  int bbat = rowm / T;
  int tt   = rowm - bbat * T;
  int kbA = (lane >> 4) << 3;     // 0 / 8
  int kbB = (lane >> 4) << 3;     // 0 / 8

  int kSteps  = K >> 5;
  int nChunks = taps * kSteps;

  auto stage = [&](int chunk, int buf) {
    int tap = chunk / kSteps;
    int k0  = (chunk - tap * kSteps) << 5;
    const _Float16* src = Bw + ((size_t)tap * Npad + n0) * K + k0;
#if HAS_TDM
    if (wv == 0)
      tdm_load_b_tile(src, (unsigned)(size_t)(void*)&bt[buf][0], K, Npad);
#else
    int r = threadIdx.x >> 2, c = (threadIdx.x & 3) * 8;   // 64 rows x 32 cols
    *(v8h*)&bt[buf][r * 32 + c] = *(const v8h*)(src + (size_t)r * K + c);
#endif
  };
  auto stage_wait = [&]() {
#if HAS_TDM
    if (wv == 0) __builtin_amdgcn_s_wait_tensorcnt(0);
#endif
  };

  v8f acc[4] = {{}, {}, {}, {}};
  v16h zz = {};

  stage(0, 0);
  stage_wait();
  __syncthreads();

  for (int ch = 0; ch < nChunks; ++ch) {
    int buf = ch & 1;
    if (ch + 1 < nChunks) stage(ch + 1, buf ^ 1);

    int tap = ch / kSteps;
    int k0  = (ch - tap * kSteps) << 5;
    int ts = tt + tap - pad;
    bool valid = (ts >= 0) && (ts < T);
    int tsc = valid ? ts : 0;                       // clamp: load stays in-bounds
    const _Float16* Ar = A + (size_t)(bbat * T + tsc) * K + k0;
    v8h a0 = *(const v8h*)(Ar + kbA);
    v8h a1 = *(const v8h*)(Ar + 16 + kbA);
    v16h av = __builtin_shufflevector(a0, a1, 0, 1, 2, 3, 4, 5, 6, 7,
                                      8, 9, 10, 11, 12, 13, 14, 15);
    av = valid ? av : zz;                           // per-lane zero for conv padding

#pragma unroll
    for (int j = 0; j < 4; ++j) {
      const _Float16* q = &bt[buf][((j * 16 + (lane & 15)) * 32) + kbB];
      v8h b0 = *(const v8h*)(q);
      v8h b1 = *(const v8h*)(q + 16);
      v16h bv = __builtin_shufflevector(b0, b1, 0, 1, 2, 3, 4, 5, 6, 7,
                                        8, 9, 10, 11, 12, 13, 14, 15);
      acc[j] = __builtin_amdgcn_wmma_f32_16x16x32_f16(false, av, false, bv,
                                                      (short)0, acc[j], false, false);
    }
    if (ch + 1 < nChunks) stage_wait();
    __syncthreads();
  }

  int mb = (lane >> 4) << 3;
#pragma unroll
  for (int j = 0; j < 4; ++j) {
    int n = n0 + j * 16 + (lane & 15);
    if (n < Nvalid) {
      float bsum = (bias ? bias[n] : 0.0f) + (bias2 ? bias2[n] : 0.0f);
#pragma unroll
      for (int r = 0; r < 8; ++r)
        Y[(size_t)(m0 + mb + r) * ldy + colOff + n] = acc[j][r] + bsum;
    }
  }
}

// ---------------------------------------------------------------------------
// One recurrent step: g[b,n] = xg[b*T+t, n] + sum_k h[b,k] * Whh[n,k]  (f16 in)
// ---------------------------------------------------------------------------
__global__ void __launch_bounds__(256)
lstm_step_gemm(const float* __restrict__ xg,
               const _Float16* __restrict__ h,
               const _Float16* __restrict__ Whh,
               float* __restrict__ g,
               int t)
{
  int wave = (int)((blockIdx.x * blockDim.x + threadIdx.x) >> 5);
  int lane = threadIdx.x & 31;
  const int ntilesN = (4 * HH) / 16;  // 32
  int mTile = wave / ntilesN;
  int nTile = wave - mTile * ntilesN;
  int m0 = mTile << 4;
  int n0 = nTile << 4;
  int n  = n0 + (lane & 15);
  int mb = (lane >> 4) << 3;
  int rowm = m0 + (lane & 15);
  int kbA = (lane >> 4) << 3;
  int kbB = (lane >> 4) << 3;

  v8f acc;
#pragma unroll
  for (int r = 0; r < 8; ++r)
    acc[r] = xg[((size_t)(m0 + mb + r) * TT + t) * (4 * HH) + n];

#pragma unroll
  for (int k0 = 0; k0 < HH; k0 += 32) {
    const _Float16* Ar = h + (size_t)rowm * HH + k0;
    v8h a0 = *(const v8h*)(Ar + kbA);
    v8h a1 = *(const v8h*)(Ar + 16 + kbA);
    v16h av = __builtin_shufflevector(a0, a1, 0, 1, 2, 3, 4, 5, 6, 7,
                                      8, 9, 10, 11, 12, 13, 14, 15);
    const _Float16* Br = Whh + (size_t)n * HH + k0;
    v8h b0 = *(const v8h*)(Br + kbB);
    v8h b1 = *(const v8h*)(Br + kbB + 16);
    v16h bv = __builtin_shufflevector(b0, b1, 0, 1, 2, 3, 4, 5, 6, 7,
                                      8, 9, 10, 11, 12, 13, 14, 15);
    acc = __builtin_amdgcn_wmma_f32_16x16x32_f16(false, av, false, bv,
                                                 (short)0, acc, false, false);
  }
#pragma unroll
  for (int r = 0; r < 8; ++r)
    g[(size_t)(m0 + mb + r) * (4 * HH) + n] = acc[r];
}

__global__ void lstm_cell(const float* __restrict__ g,
                          _Float16* __restrict__ h,
                          float* __restrict__ c,
                          _Float16* __restrict__ lstm_out_h,
                          int t, int dirOff)
{
  int idx = blockIdx.x * blockDim.x + threadIdx.x;
  if (idx >= BB * HH) return;
  int b = idx >> 7;
  int u = idx & (HH - 1);
  const float* gr = g + (size_t)b * 4 * HH;
  float gi = gr[u], gf = gr[HH + u], gg = gr[2 * HH + u], go = gr[3 * HH + u];
  float cc = sigmoidf_(gf) * c[idx] + sigmoidf_(gi) * tanhf(gg);
  float hh2 = sigmoidf_(go) * tanhf(cc);
  c[idx] = cc;
  h[idx] = (_Float16)hh2;
  lstm_out_h[((size_t)b * TT + t) * (2 * HH) + dirOff + u] = (_Float16)hh2;
}

// ---------------------------------------------------------------------------
// Pre-pack helpers
// ---------------------------------------------------------------------------
__global__ void cvt_f16(const float* __restrict__ x, _Float16* __restrict__ y, size_t n)
{
  size_t i = (size_t)blockIdx.x * blockDim.x + threadIdx.x;
  if (i < n) y[i] = (_Float16)x[i];
}

// w[n][k][tap] f32 -> o[tap][Npad][k] f16 (rows n>=N zero-filled)
__global__ void repack_w(const float* __restrict__ w, _Float16* __restrict__ o,
                         int N, int K, int taps, int Npad)
{
  size_t idx = (size_t)blockIdx.x * blockDim.x + threadIdx.x;
  size_t tot = (size_t)taps * Npad * K;
  if (idx >= tot) return;
  int k = (int)(idx % K);
  size_t r = idx / K;
  int n = (int)(r % Npad);
  int tap = (int)(r / Npad);
  float v = (n < N) ? w[((size_t)n * K + k) * taps + tap] : 0.0f;
  o[idx] = (_Float16)v;
}

// ---------------------------------------------------------------------------
// BatchNorm stats (biased var) — one block per channel.
// ---------------------------------------------------------------------------
__global__ void bn_stats(const float* __restrict__ x, int rows, int cols,
                         float* __restrict__ meanvar)
{
  __shared__ float ss[256], sq[256];
  int ch = blockIdx.x;
  float s = 0.f, s2 = 0.f;
  for (int r = threadIdx.x; r < rows; r += 256) {
    float v = x[(size_t)r * cols + ch];
    s += v; s2 += v * v;
  }
  ss[threadIdx.x] = s; sq[threadIdx.x] = s2;
  __syncthreads();
  for (int st = 128; st > 0; st >>= 1) {
    if (threadIdx.x < st) {
      ss[threadIdx.x] += ss[threadIdx.x + st];
      sq[threadIdx.x] += sq[threadIdx.x + st];
    }
    __syncthreads();
  }
  if (threadIdx.x == 0) {
    float m = ss[0] / (float)rows;
    meanvar[ch] = m;
    meanvar[cols + ch] = sq[0] / (float)rows - m * m;
  }
}

// normalize + GELU, emit f16 for downstream WMMA consumers
__global__ void bn_gelu_h(const float* __restrict__ x, _Float16* __restrict__ xh,
                          int rows, int cols,
                          const float* __restrict__ meanvar,
                          const float* __restrict__ gw,
                          const float* __restrict__ bw)
{
  size_t idx = (size_t)blockIdx.x * blockDim.x + threadIdx.x;
  if (idx >= (size_t)rows * cols) return;
  int ch = (int)(idx % cols);
  float m = meanvar[ch], v = meanvar[cols + ch];
  float y = (x[idx] - m) * rsqrtf(v + 1e-5f) * gw[ch] + bw[ch];
  xh[idx] = (_Float16)gelu_(y);
}

// same but keep f32 (for bn2 -> needed f32? classifier takes f16; keep f16 only)

// ---------------------------------------------------------------------------
// CRF + CE (unchanged from previous round)
// ---------------------------------------------------------------------------
__global__ void crf_partition_k(const float* __restrict__ logits,
                                const int* __restrict__ amask,
                                const long long* __restrict__ labels,
                                const float* __restrict__ start,
                                const float* __restrict__ trans,
                                const float* __restrict__ endv,
                                float* __restrict__ logZ)
{
  __shared__ float tr[KK * KK];
  if (threadIdx.x < KK * KK) tr[threadIdx.x] = trans[threadIdx.x];
  __syncthreads();
  int b = blockIdx.x * blockDim.x + threadIdx.x;
  if (b >= BB) return;
  float sc[KK];
#pragma unroll
  for (int j = 0; j < KK; ++j) sc[j] = start[j] + logits[(size_t)b * TT * KK + j];
  for (int t = 1; t < TT; ++t) {
    const float* em = logits + ((size_t)b * TT + t) * KK;
    bool m = (amask[b * TT + t] != 0) && (labels[b * TT + t] != -100LL);
    float nxt[KK];
#pragma unroll
    for (int j = 0; j < KK; ++j) {
      float mx = -3.0e38f;
#pragma unroll
      for (int i = 0; i < KK; ++i) mx = fmaxf(mx, sc[i] + tr[i * KK + j]);
      float s = 0.f;
#pragma unroll
      for (int i = 0; i < KK; ++i) s += expf(sc[i] + tr[i * KK + j] - mx);
      nxt[j] = mx + logf(s) + em[j];
    }
#pragma unroll
    for (int j = 0; j < KK; ++j) sc[j] = m ? nxt[j] : sc[j];
  }
  float mx = -3.0e38f;
#pragma unroll
  for (int j = 0; j < KK; ++j) mx = fmaxf(mx, sc[j] + endv[j]);
  float s = 0.f;
#pragma unroll
  for (int j = 0; j < KK; ++j) s += expf(sc[j] + endv[j] - mx);
  logZ[b] = mx + logf(s);
}

__global__ void crf_numerator_k(const float* __restrict__ logits,
                                const int* __restrict__ amask,
                                const long long* __restrict__ labels,
                                const float* __restrict__ start,
                                const float* __restrict__ trans,
                                const float* __restrict__ endv,
                                float* __restrict__ num)
{
  int b = blockIdx.x * blockDim.x + threadIdx.x;
  if (b >= BB) return;
  const long long* lb = labels + (size_t)b * TT;
  const int* am = amask + (size_t)b * TT;
  const float* em = logits + (size_t)b * TT * KK;
  long long l0 = lb[0];
  int tag0 = (l0 == -100LL) ? 0 : (int)l0;
  float acc = start[tag0] + em[tag0];
  int prev = tag0;
  int cnt = ((am[0] != 0) && (l0 != -100LL)) ? 1 : 0;
  for (int t = 1; t < TT; ++t) {
    long long lt = lb[t];
    int tg = (lt == -100LL) ? 0 : (int)lt;
    bool m = (am[t] != 0) && (lt != -100LL);
    float mf = m ? 1.0f : 0.0f;
    acc += mf * (trans[prev * KK + tg] + em[t * KK + tg]);
    prev = tg;
    cnt += m ? 1 : 0;
  }
  int se = cnt - 1;
  if (se < 0) se = 0;
  if (se >= TT) se = TT - 1;
  long long ll = lb[se];
  int lastTag = (ll == -100LL) ? 0 : (int)ll;
  acc += endv[lastTag];
  num[b] = acc;
}

__global__ void crf_reduce_k(const float* __restrict__ num,
                             const float* __restrict__ logZ,
                             float* __restrict__ acc)
{
  __shared__ float sm[256];
  int t = threadIdx.x;
  sm[t] = num[t] - logZ[t];
  __syncthreads();
  for (int st = 128; st > 0; st >>= 1) {
    if (t < st) sm[t] += sm[t + st];
    __syncthreads();
  }
  if (t == 0) acc[0] = sm[0];
}

__global__ void ce_kernel(const float* __restrict__ logits,
                          const long long* __restrict__ labels,
                          const float* __restrict__ cw,
                          float* __restrict__ acc)
{
  __shared__ float s0[256], s1[256];
  int i = blockIdx.x * blockDim.x + threadIdx.x;
  float wn = 0.f, wv2 = 0.f;
  if (i < BB * TT) {
    float l[KK];
    float mx = -3.0e38f;
#pragma unroll
    for (int j = 0; j < KK; ++j) { l[j] = logits[(size_t)i * KK + j]; mx = fmaxf(mx, l[j]); }
    float s = 0.f;
#pragma unroll
    for (int j = 0; j < KK; ++j) s += expf(l[j] - mx);
    float lse = mx + logf(s);
    long long y = labels[i];
    bool valid = (y != -100LL);
    int yi = valid ? (int)y : 0;
    float ly = l[0];
#pragma unroll
    for (int j = 1; j < KK; ++j) ly = (yi == j) ? l[j] : ly;
    float w = valid ? cw[yi] : 0.f;
    wn = w * (lse - ly);
    wv2 = w;
  }
  s0[threadIdx.x] = wn; s1[threadIdx.x] = wv2;
  __syncthreads();
  for (int st = 128; st > 0; st >>= 1) {
    if (threadIdx.x < st) {
      s0[threadIdx.x] += s0[threadIdx.x + st];
      s1[threadIdx.x] += s1[threadIdx.x + st];
    }
    __syncthreads();
  }
  if (threadIdx.x == 0) {
    atomicAdd(&acc[1], s0[0]);
    atomicAdd(&acc[2], s1[0]);
  }
}

__global__ void final_combine_k(const float* __restrict__ acc, float* __restrict__ out)
{
  if (threadIdx.x == 0 && blockIdx.x == 0) {
    float crf = -(acc[0] / (float)BB);
    float ce  = acc[1] / acc[2];
    out[0] = 0.8f * crf + 0.2f * ce;
  }
}

// ---------------------------------------------------------------------------
extern "C" void kernel_launch(void* const* d_in, const int* in_sizes, int n_in,
                              void* d_out, int out_size, void* d_ws, size_t ws_size,
                              hipStream_t stream)
{
  const float*     emb      = (const float*)d_in[0];
  const int*       amask    = (const int*)d_in[1];
  const long long* labels   = (const long long*)d_in[2];
  const float*     conv5_w  = (const float*)d_in[3];
  const float*     conv5_b  = (const float*)d_in[4];
  const float*     bn1_g    = (const float*)d_in[5];
  const float*     bn1_b    = (const float*)d_in[6];
  const float*     Wih_f    = (const float*)d_in[7];
  const float*     Whh_f    = (const float*)d_in[8];
  const float*     bih_f    = (const float*)d_in[9];
  const float*     bhh_f    = (const float*)d_in[10];
  const float*     Wih_b    = (const float*)d_in[11];
  const float*     Whh_b    = (const float*)d_in[12];
  const float*     bih_b    = (const float*)d_in[13];
  const float*     bhh_b    = (const float*)d_in[14];
  const float*     conv7_w  = (const float*)d_in[15];
  const float*     conv7_b  = (const float*)d_in[16];
  const float*     conv9_w  = (const float*)d_in[17];
  const float*     conv9_b  = (const float*)d_in[18];
  const float*     bn2_g    = (const float*)d_in[19];
  const float*     bn2_b    = (const float*)d_in[20];
  const float*     cls_w    = (const float*)d_in[21];
  const float*     cls_b    = (const float*)d_in[22];
  const float*     crf_start= (const float*)d_in[23];
  const float*     crf_trans= (const float*)d_in[24];
  const float*     crf_end  = (const float*)d_in[25];
  const float*     cweights = (const float*)d_in[26];
  (void)in_sizes; (void)n_in; (void)out_size; (void)ws_size;

  char* ws = (char*)d_ws;
  size_t off = 0;
  auto carve = [&](size_t bytes) -> char* {
    char* p = ws + off;
    off += (bytes + 255) & ~(size_t)255;
    return p;
  };
  _Float16* embh   = (_Float16*)carve((size_t)BB * TT * DD * 2);        // 67 MB
  float*    xc     = (float*)carve((size_t)BB * TT * CFF * 4);          // 16.8 MB
  _Float16* xch    = (_Float16*)carve((size_t)BB * TT * CFF * 2);       // 8.4 MB
  float*    gates  = (float*)carve((size_t)BB * TT * 4 * HH * 4);       // 67 MB
  _Float16* louth  = (_Float16*)carve((size_t)BB * TT * 2 * HH * 2);    // 16.8 MB
  float*    feat   = (float*)carve((size_t)BB * TT * 2 * CFF * 4);      // 33.5 MB
  _Float16* feath  = (_Float16*)carve((size_t)BB * TT * 2 * CFF * 2);   // 16.8 MB
  float*    logits = (float*)carve((size_t)BB * TT * KK * 4);
  _Float16* hbufh  = (_Float16*)carve((size_t)BB * HH * 2);
  float*    cbuf   = (float*)carve((size_t)BB * HH * 4);
  float*    gcur   = (float*)carve((size_t)BB * 4 * HH * 4);
  _Float16* c5h    = (_Float16*)carve((size_t)5 * CFF * DD * 2);
  _Float16* c7h    = (_Float16*)carve((size_t)7 * CFF * 2 * HH * 2);
  _Float16* c9h    = (_Float16*)carve((size_t)9 * CFF * 2 * HH * 2);
  _Float16* wihf   = (_Float16*)carve((size_t)4 * HH * CFF * 2);
  _Float16* wihb   = (_Float16*)carve((size_t)4 * HH * CFF * 2);
  _Float16* whhf   = (_Float16*)carve((size_t)4 * HH * HH * 2);
  _Float16* whhb   = (_Float16*)carve((size_t)4 * HH * HH * 2);
  _Float16* clsh   = (_Float16*)carve((size_t)64 * 2 * CFF * 2);
  float*    meanvar= (float*)carve((size_t)2 * 2 * CFF * 4);
  float*    logZ   = (float*)carve((size_t)BB * 4);
  float*    numb   = (float*)carve((size_t)BB * 4);
  float*    accum  = (float*)carve(256);

  hipMemsetAsync(accum, 0, 256, stream);

  // ---- pre-pack activations/weights to f16 ----
  {
    size_t n = (size_t)BB * TT * DD;
    cvt_f16<<<(unsigned)((n + 255) / 256), 256, 0, stream>>>(emb, embh, n);
    auto rp = [&](const float* w, _Float16* o, int N, int K, int taps, int Npad) {
      size_t tot = (size_t)taps * Npad * K;
      repack_w<<<(unsigned)((tot + 255) / 256), 256, 0, stream>>>(w, o, N, K, taps, Npad);
    };
    rp(conv5_w, c5h, CFF, DD, 5, CFF);
    rp(conv7_w, c7h, CFF, 2 * HH, 7, CFF);
    rp(conv9_w, c9h, CFF, 2 * HH, 9, CFF);
    rp(Wih_f, wihf, 4 * HH, CFF, 1, 4 * HH);
    rp(Wih_b, wihb, 4 * HH, CFF, 1, 4 * HH);
    rp(Whh_f, whhf, 4 * HH, HH, 1, 4 * HH);
    rp(Whh_b, whhb, 4 * HH, HH, 1, 4 * HH);
    rp(cls_w, clsh, KK, 2 * CFF, 1, 64);
  }

  const int mBlocks = (BB * TT) / 128;   // 256

  // conv5: M=32768 N=128 K=5*1024 -> xc
  wmma_gemm_tdm<<<mBlocks * 2, 256, 0, stream>>>(embh, c5h, conv5_b, nullptr,
      xc, TT, DD, 5, 2, CFF, 2, CFF, 0, CFF);
  bn_stats<<<CFF, 256, 0, stream>>>(xc, BB * TT, CFF, meanvar);
  bn_gelu_h<<<(BB * TT * CFF) / 256, 256, 0, stream>>>(xc, xch, BB * TT, CFF,
                                                       meanvar, bn1_g, bn1_b);

  // ---- forward LSTM ----
  wmma_gemm_tdm<<<mBlocks * 8, 256, 0, stream>>>(xch, wihf, bih_f, bhh_f,
      gates, TT, CFF, 1, 0, 4 * HH, 8, 4 * HH, 0, 4 * HH);
  hipMemsetAsync(hbufh, 0, (size_t)BB * HH * 2, stream);
  hipMemsetAsync(cbuf, 0, (size_t)BB * HH * 4, stream);
  {
    int stepBlocks = ((BB / 16) * ((4 * HH) / 16)) / 8;  // 64
    for (int t = 0; t < TT; ++t) {
      lstm_step_gemm<<<stepBlocks, 256, 0, stream>>>(gates, hbufh, whhf, gcur, t);
      lstm_cell<<<(BB * HH) / 256, 256, 0, stream>>>(gcur, hbufh, cbuf, louth, t, 0);
    }
  }
  // ---- backward LSTM ----
  wmma_gemm_tdm<<<mBlocks * 8, 256, 0, stream>>>(xch, wihb, bih_b, bhh_b,
      gates, TT, CFF, 1, 0, 4 * HH, 8, 4 * HH, 0, 4 * HH);
  hipMemsetAsync(hbufh, 0, (size_t)BB * HH * 2, stream);
  hipMemsetAsync(cbuf, 0, (size_t)BB * HH * 4, stream);
  {
    int stepBlocks = ((BB / 16) * ((4 * HH) / 16)) / 8;
    for (int t = TT - 1; t >= 0; --t) {
      lstm_step_gemm<<<stepBlocks, 256, 0, stream>>>(gates, hbufh, whhb, gcur, t);
      lstm_cell<<<(BB * HH) / 256, 256, 0, stream>>>(gcur, hbufh, cbuf, louth, t, HH);
    }
  }

  // conv7 / conv9 -> feat (channel concat at col 0 / 128)
  wmma_gemm_tdm<<<mBlocks * 2, 256, 0, stream>>>(louth, c7h, conv7_b, nullptr,
      feat, TT, 2 * HH, 7, 3, CFF, 2, 2 * CFF, 0, CFF);
  wmma_gemm_tdm<<<mBlocks * 2, 256, 0, stream>>>(louth, c9h, conv9_b, nullptr,
      feat, TT, 2 * HH, 9, 4, CFF, 2, 2 * CFF, CFF, CFF);
  bn_stats<<<2 * CFF, 256, 0, stream>>>(feat, BB * TT, 2 * CFF, meanvar);
  bn_gelu_h<<<(BB * TT * 2 * CFF) / 256, 256, 0, stream>>>(feat, feath, BB * TT, 2 * CFF,
                                                           meanvar, bn2_g, bn2_b);

  // classifier: N=9 (padded to 64), single n-block
  wmma_gemm_tdm<<<mBlocks * 1, 256, 0, stream>>>(feath, clsh, cls_b, nullptr,
      logits, TT, 2 * CFF, 1, 0, KK, 1, KK, 0, 64);

  crf_partition_k<<<1, 256, 0, stream>>>(logits, amask, labels, crf_start, crf_trans, crf_end, logZ);
  crf_numerator_k<<<1, 256, 0, stream>>>(logits, amask, labels, crf_start, crf_trans, crf_end, numb);
  crf_reduce_k<<<1, 256, 0, stream>>>(numb, logZ, accum);
  ce_kernel<<<(BB * TT) / 256, 256, 0, stream>>>(logits, labels, cweights, accum);
  final_combine_k<<<1, 32, 0, stream>>>(accum, (float*)d_out);
}
